// HybridTemporalGNN_84207128805739
// MI455X (gfx1250) — matrix operand order repaired and checked by reading, not compiled, not run on hardware
//
#include <hip/hip_runtime.h>
#include <hip/hip_bf16.h>
#include <math.h>

// ---------------------------------------------------------------------------
// HybridTemporalGNN for MI455X (gfx1250): bf16 WMMA GEMMs + f32 atomics GAT
// GEMM register-blocked 1 M-tile x 4 N-tiles per wave (A-fragment reuse).
// ---------------------------------------------------------------------------

typedef __bf16 bf16;
typedef __attribute__((ext_vector_type(16))) __bf16 v16bf;
typedef __attribute__((ext_vector_type(8)))  __bf16 v8bf;
typedef __attribute__((ext_vector_type(8)))  float  v8f;

#define TT 6
#define NN 2000
#define EE 2000
#define EE2 4000
#define HIDC 256
#define LSTMH 128
#define NEG_SLOPE 0.2f
#define BN_EPS 1e-5f

// ---------------- device helpers ----------------

static __device__ __forceinline__ float wave_sum(float v) {
#pragma unroll
  for (int off = 16; off > 0; off >>= 1) v += __shfl_xor(v, off, 32);
  return v;
}

// monotone float -> uint key (order-preserving incl. negatives)
static __device__ __forceinline__ unsigned fkey(float f) {
  unsigned b = __float_as_uint(f);
  return (b & 0x80000000u) ? ~b : (b | 0x80000000u);
}
static __device__ __forceinline__ float fkey_inv(unsigned k) {
  unsigned b = (k & 0x80000000u) ? (k & 0x7fffffffu) : ~k;
  return __uint_as_float(b);
}

static __device__ __forceinline__ float sigmoidf(float x) {
  return 1.f / (1.f + expf(-x));
}

// ---------------- utility kernels ----------------

__global__ void k_zero(float* __restrict__ p, int n) {
  int i = blockIdx.x * blockDim.x + threadIdx.x;
  if (i < n) p[i] = 0.f;
}

// build edge lists with self-loops appended
__global__ void k_build_edges(const int* __restrict__ eidx,
                              int* __restrict__ src2, int* __restrict__ dst2) {
  int e = blockIdx.x * blockDim.x + threadIdx.x;
  if (e >= EE2) return;
  if (e < EE) { src2[e] = eidx[e]; dst2[e] = eidx[EE + e]; }
  else        { src2[e] = e - EE;  dst2[e] = e - EE; }
}

// per-node degree + edge_attr sum (for fill_value='mean' self-loop attr)
__global__ void k_deg(const int* __restrict__ dst2, const float* __restrict__ eattr_t,
                      float* __restrict__ deg, float* __restrict__ easum) {
  int e = blockIdx.x * blockDim.x + threadIdx.x;
  if (e >= EE) return;
  int d = dst2[e];
  unsafeAtomicAdd(&deg[d], 1.f);
  unsafeAtomicAdd(&easum[d], eattr_t[e]);
}

__global__ void k_ea2(const float* __restrict__ eattr_t, const float* __restrict__ deg,
                      const float* __restrict__ easum, float* __restrict__ ea2) {
  int e = blockIdx.x * blockDim.x + threadIdx.x;
  if (e >= EE2) return;
  ea2[e] = (e < EE) ? eattr_t[e] : easum[e - EE] / fmaxf(deg[e - EE], 1.f);
}

// f32 [M,K] row-major -> bf16 [M,Kp] zero-padded
__global__ void k_cvt_pad(const float* __restrict__ in, bf16* __restrict__ out,
                          int M, int K, int Kp) {
  int i = blockIdx.x * blockDim.x + threadIdx.x;
  if (i >= M * Kp) return;
  int r = i / Kp, k = i - r * Kp;
  out[i] = (k < K) ? (bf16)in[(size_t)r * K + k] : (bf16)0.f;
}

// weight f32 [K,N] -> transposed bf16 [N,Kp] zero-padded (contiguous B-fragment loads)
__global__ void k_wt(const float* __restrict__ W, bf16* __restrict__ Wt,
                     int K, int Nc, int Kp) {
  int i = blockIdx.x * blockDim.x + threadIdx.x;
  if (i >= Nc * Kp) return;
  int n = i / Kp, k = i - n * Kp;
  Wt[i] = (k < K) ? (bf16)W[(size_t)k * Nc + n] : (bf16)0.f;
}

// ---------------- WMMA GEMM: C[M,Nc] = A[M,Kp](bf16) @ Bt[Nc,Kp]^T + bias ----
// one wave per 16x64 output block (1 M-tile x 4 N-tiles); A fragment reused 4x;
// K stepped by 32 (v_wmma_f32_16x16x32_bf16, f32 accumulate).

__global__ void k_wmma_gemm(const bf16* __restrict__ A, const bf16* __restrict__ Bt,
                            const float* __restrict__ bias, float* __restrict__ C,
                            int M, int Nc, int Kp) {
  const int wave = (int)((blockIdx.x * blockDim.x + threadIdx.x) >> 5);
  const int lane = threadIdx.x & 31;
  const int tilesN4 = Nc >> 6;              // groups of 4 N-tiles
  const int tm = wave / tilesN4;
  const int tg = wave - tm * tilesN4;
  if (tm * 16 >= M) return;

  const int half = lane >> 4;               // which 16-lane half of the wave
  const int l16  = lane & 15;

  // A fragment rows: m = l16 ; A 16x32 k order used here (contiguous, same as B):
  //   elems 0..7  : K = k0 + half*16 + i
  //   elems 8..15 : K = k0 + half*16 + i
  const bf16* arow = A  + (size_t)(tm * 16 + l16) * Kp;
  // B fragment (K x N, col n = l16): elems i: K = k0 + half*16 + i -> contiguous in Bt
  const bf16* brow = Bt + (size_t)(tg * 64 + l16) * Kp + half * 16;
  const size_t bstride = (size_t)16 * Kp;   // next N-tile

  v8f acc0 = {0.f,0.f,0.f,0.f,0.f,0.f,0.f,0.f};
  v8f acc1 = acc0, acc2 = acc0, acc3 = acc0;

  for (int k0 = 0; k0 < Kp; k0 += 32) {
    if (k0 + 32 < Kp) {                     // gfx1250 global_prefetch_b8
      __builtin_prefetch(arow + k0 + 32, 0, 1);
      __builtin_prefetch(brow + k0 + 32, 0, 1);
    }
    v8bf alo = *(const v8bf*)(arow + k0 + half * 16);       // 16B load
    v8bf ahi = *(const v8bf*)(arow + k0 + half * 16 + 8);   // 16B load
    v16bf a = __builtin_shufflevector(alo, ahi,
        0, 1, 2, 3, 4, 5, 6, 7, 8, 9, 10, 11, 12, 13, 14, 15);
    v16bf b0 = *(const v16bf*)(brow + k0);                  // 32B loads
    v16bf b1 = *(const v16bf*)(brow + bstride + k0);
    v16bf b2 = *(const v16bf*)(brow + 2 * bstride + k0);
    v16bf b3 = *(const v16bf*)(brow + 3 * bstride + k0);
    acc0 = __builtin_amdgcn_wmma_f32_16x16x32_bf16(false, a, false, b0, (short)0, acc0, false, false);
    acc1 = __builtin_amdgcn_wmma_f32_16x16x32_bf16(false, a, false, b1, (short)0, acc1, false, false);
    acc2 = __builtin_amdgcn_wmma_f32_16x16x32_bf16(false, a, false, b2, (short)0, acc2, false, false);
    acc3 = __builtin_amdgcn_wmma_f32_16x16x32_bf16(false, a, false, b3, (short)0, acc3, false, false);
  }

  // C/D layout: VGPR r -> row half*8+r, col l16
  float* crow = C + (size_t)(tm * 16 + half * 8) * Nc + tg * 64 + l16;
#pragma unroll
  for (int j = 0; j < 4; ++j) {
    const int col = tg * 64 + j * 16 + l16;
    const float bv = bias ? bias[col] : 0.f;
    const v8f acc = (j == 0) ? acc0 : (j == 1) ? acc1 : (j == 2) ? acc2 : acc3;
#pragma unroll
    for (int r = 0; r < 8; ++r) crow[(size_t)r * Nc + j * 16] = acc[r] + bv;
  }
}

// ---------------- GATv2 edge kernels ----------------
// one wave per edge; channels strided across lanes (ch is a multiple of 32)

__global__ void k_gat_alpha(const float* __restrict__ xp, const int* __restrict__ src2,
                            const int* __restrict__ dst2, const float* __restrict__ ea2,
                            const float* __restrict__ We, const float* __restrict__ att,
                            float* __restrict__ alpha, unsigned* __restrict__ amaxKey,
                            int heads, int ch) {
  const int e = (int)((blockIdx.x * blockDim.x + threadIdx.x) >> 5);
  const int lane = threadIdx.x & 31;
  if (e >= EE2) return;
  const int s = src2[e], d = dst2[e];
  const float eav = ea2[e];
  const float* xs = xp + (size_t)s * HIDC;
  const float* xd = xp + (size_t)d * HIDC;
  const int chunksPerHead = ch >> 5;
  float hsum = 0.f;
  int chunk = 0, h = 0;
  for (int j = 0; j < (HIDC >> 5); ++j) {
    const int c = (j << 5) + lane;
    float m = xs[c] + xd[c] + eav * We[c];
    float g = m > 0.f ? m : NEG_SLOPE * m;        // leaky_relu
    hsum += wave_sum(g * att[c]);
    if (++chunk == chunksPerHead) {
      if (lane == 0) {
        alpha[(size_t)e * heads + h] = hsum;
        atomicMax(&amaxKey[(size_t)d * heads + h], fkey(hsum));
      }
      hsum = 0.f; chunk = 0; ++h;
    }
  }
}

__global__ void k_gat_soft(float* __restrict__ alpha, const unsigned* __restrict__ amaxKey,
                           const int* __restrict__ dst2, float* __restrict__ denom,
                           int heads) {
  int i = blockIdx.x * blockDim.x + threadIdx.x;
  if (i >= EE2 * heads) return;
  int e = i / heads, h = i - e * heads;
  int d = dst2[e];
  float ex = expf(alpha[i] - fkey_inv(amaxKey[(size_t)d * heads + h]));
  alpha[i] = ex;
  unsafeAtomicAdd(&denom[(size_t)d * heads + h], ex);
}

__global__ void k_gat_agg(const float* __restrict__ xp, const float* __restrict__ alpha,
                          const float* __restrict__ denom, const int* __restrict__ src2,
                          const int* __restrict__ dst2, float* __restrict__ outb,
                          int heads, int ch) {
  const int e = (int)((blockIdx.x * blockDim.x + threadIdx.x) >> 5);
  const int lane = threadIdx.x & 31;
  if (e >= EE2) return;
  const int s = src2[e], d = dst2[e];
  for (int c = lane; c < HIDC; c += 32) {
    int h = c / ch;
    float a = alpha[(size_t)e * heads + h] / denom[(size_t)d * heads + h];
    unsafeAtomicAdd(&outb[(size_t)d * HIDC + c], a * xp[(size_t)s * HIDC + c]);
  }
}

// out+bias -> eval BatchNorm -> ELU -> bf16 (input of next WMMA GEMM)
__global__ void k_post(const float* __restrict__ outb, const float* __restrict__ bias,
                       const float* __restrict__ gamma, const float* __restrict__ beta,
                       const float* __restrict__ mean, const float* __restrict__ var,
                       bf16* __restrict__ Abf) {
  int i = blockIdx.x * blockDim.x + threadIdx.x;
  if (i >= NN * HIDC) return;
  int c = i & (HIDC - 1);
  float v = outb[i] + bias[c];
  v = (v - mean[c]) * rsqrtf(var[c] + BN_EPS) * gamma[c] + beta[c];
  v = v > 0.f ? v : (expf(v) - 1.f);              // ELU
  Abf[i] = (bf16)v;
}

// global mean pool over nodes (+conv3 bias): one block, one channel per thread
__global__ void k_pool(const float* __restrict__ outb, const float* __restrict__ bias,
                       float* __restrict__ grepr_t) {
  int c = threadIdx.x;   // 0..255
  float s = 0.f;
  for (int n = 0; n < NN; ++n) s += outb[(size_t)n * HIDC + c];
  grepr_t[c] = s * (1.f / (float)NN) + bias[c];
}

// ---------------- 2-layer LSTM + FC, one 512-thread block ----------------

__global__ void k_lstm_fc(const float* __restrict__ grepr,
                          const float* __restrict__ Wih0, const float* __restrict__ Whh0,
                          const float* __restrict__ bih0, const float* __restrict__ bhh0,
                          const float* __restrict__ Wih1, const float* __restrict__ Whh1,
                          const float* __restrict__ bih1, const float* __restrict__ bhh1,
                          const float* __restrict__ fcW, const float* __restrict__ fcb,
                          float* __restrict__ out) {
  __shared__ float h[LSTMH], c[LSTMH], gates[4 * LSTMH], hs[TT * LSTMH], red[LSTMH];
  const int tid = threadIdx.x;   // 0..511, one gate row each
  if (tid < LSTMH) { h[tid] = 0.f; c[tid] = 0.f; }
  __syncthreads();

  // layer 0: input HIDC
  for (int t = 0; t < TT; ++t) {
    float acc = bih0[tid] + bhh0[tid];
    const float* wr = Wih0 + (size_t)tid * HIDC;
    const float* xr = grepr + (size_t)t * HIDC;
    for (int k = 0; k < HIDC; ++k) acc += wr[k] * xr[k];
    const float* wh = Whh0 + (size_t)tid * LSTMH;
    for (int k = 0; k < LSTMH; ++k) acc += wh[k] * h[k];
    gates[tid] = acc;
    __syncthreads();
    if (tid < LSTMH) {
      float ig = sigmoidf(gates[tid]);
      float fg = sigmoidf(gates[LSTMH + tid]);
      float gg = tanhf(gates[2 * LSTMH + tid]);
      float og = sigmoidf(gates[3 * LSTMH + tid]);
      float cn = fg * c[tid] + ig * gg;
      c[tid] = cn;
      float hn = og * tanhf(cn);
      h[tid] = hn;
      hs[t * LSTMH + tid] = hn;
    }
    __syncthreads();
  }

  if (tid < LSTMH) { h[tid] = 0.f; c[tid] = 0.f; }
  __syncthreads();

  // layer 1: input LSTMH
  for (int t = 0; t < TT; ++t) {
    float acc = bih1[tid] + bhh1[tid];
    const float* wr = Wih1 + (size_t)tid * LSTMH;
    const float* xr = hs + (size_t)t * LSTMH;
    for (int k = 0; k < LSTMH; ++k) acc += wr[k] * xr[k];
    const float* wh = Whh1 + (size_t)tid * LSTMH;
    for (int k = 0; k < LSTMH; ++k) acc += wh[k] * h[k];
    gates[tid] = acc;
    __syncthreads();
    if (tid < LSTMH) {
      float ig = sigmoidf(gates[tid]);
      float fg = sigmoidf(gates[LSTMH + tid]);
      float gg = tanhf(gates[2 * LSTMH + tid]);
      float og = sigmoidf(gates[3 * LSTMH + tid]);
      float cn = fg * c[tid] + ig * gg;
      c[tid] = cn;
      h[tid] = og * tanhf(cn);
    }
    __syncthreads();
  }

  // FC: out = h @ fcW^T + fcb
  if (tid < LSTMH) red[tid] = fcW[tid] * h[tid];
  __syncthreads();
  if (tid == 0) {
    float s = fcb[0];
    for (int k = 0; k < LSTMH; ++k) s += red[k];
    out[0] = s;
  }
}

// ---------------------------------------------------------------------------
// host-side orchestration
// ---------------------------------------------------------------------------

extern "C" void kernel_launch(void* const* d_in, const int* in_sizes, int n_in,
                              void* d_out, int out_size, void* d_ws, size_t ws_size,
                              hipStream_t stream) {
  (void)in_sizes; (void)n_in; (void)out_size; (void)ws_size;

  // inputs in setup_inputs() insertion order (nested dict insertion order)
  const float* x_seq = (const float*)d_in[0];    // [T,N,31]
  const int*   eidx  = (const int*)d_in[1];      // [2,E]
  const float* eattr = (const float*)d_in[2];    // [T,E,1]
  struct Conv { const float *Wl, *bl, *We, *att, *bias; };
  Conv cv[3];
  for (int i = 0; i < 3; ++i) {
    cv[i].Wl   = (const float*)d_in[3 + 5 * i + 0];
    cv[i].bl   = (const float*)d_in[3 + 5 * i + 1];
    cv[i].We   = (const float*)d_in[3 + 5 * i + 2];
    cv[i].att  = (const float*)d_in[3 + 5 * i + 3];
    cv[i].bias = (const float*)d_in[3 + 5 * i + 4];
  }
  const float* bn_g[2] = {(const float*)d_in[18], (const float*)d_in[22]};
  const float* bn_b[2] = {(const float*)d_in[19], (const float*)d_in[23]};
  const float* bn_m[2] = {(const float*)d_in[20], (const float*)d_in[24]};
  const float* bn_v[2] = {(const float*)d_in[21], (const float*)d_in[25]};
  const float* Wih0 = (const float*)d_in[26];
  const float* Whh0 = (const float*)d_in[27];
  const float* bih0 = (const float*)d_in[28];
  const float* bhh0 = (const float*)d_in[29];
  const float* Wih1 = (const float*)d_in[30];
  const float* Whh1 = (const float*)d_in[31];
  const float* bih1 = (const float*)d_in[32];
  const float* bhh1 = (const float*)d_in[33];
  const float* fcW  = (const float*)d_in[34];
  const float* fcb  = (const float*)d_in[35];

  // workspace bump allocator (256B aligned)
  char* ws = (char*)d_ws;
  size_t off = 0;
  auto alloc = [&](size_t bytes) -> void* {
    void* p = ws + off;
    off += (bytes + 255) & ~(size_t)255;
    return p;
  };
  int*      src2    = (int*)alloc(EE2 * 4);
  int*      dst2    = (int*)alloc(EE2 * 4);
  float*    ea2     = (float*)alloc(EE2 * 4);
  float*    deg     = (float*)alloc(NN * 4);
  float*    easum   = (float*)alloc(NN * 4);
  bf16*     Wt1     = (bf16*)alloc((size_t)HIDC * 32 * 2);
  bf16*     Wt2     = (bf16*)alloc((size_t)HIDC * HIDC * 2);
  bf16*     Wt3     = (bf16*)alloc((size_t)HIDC * HIDC * 2);
  bf16*     Abf     = (bf16*)alloc((size_t)NN * HIDC * 2);
  float*    xp      = (float*)alloc((size_t)NN * HIDC * 4);
  float*    outb    = (float*)alloc((size_t)NN * HIDC * 4);
  float*    alphaB  = (float*)alloc((size_t)EE2 * 8 * 4);
  unsigned* amaxKey = (unsigned*)alloc((size_t)NN * 8 * 4);
  float*    denom   = (float*)alloc((size_t)NN * 8 * 4);
  float*    grepr   = (float*)alloc((size_t)TT * HIDC * 4);

  const int B = 256;
  auto blks = [](int n, int b) { return (n + b - 1) / b; };

  // once-per-launch prep
  k_build_edges<<<blks(EE2, B), B, 0, stream>>>(eidx, src2, dst2);
  k_wt<<<blks(HIDC * 32, B),   B, 0, stream>>>(cv[0].Wl, Wt1, 31,   HIDC, 32);
  k_wt<<<blks(HIDC * HIDC, B), B, 0, stream>>>(cv[1].Wl, Wt2, HIDC, HIDC, HIDC);
  k_wt<<<blks(HIDC * HIDC, B), B, 0, stream>>>(cv[2].Wl, Wt3, HIDC, HIDC, HIDC);

  const int gemmWaves  = (NN / 16) * (HIDC / 64);        // 125 x 4 = 500 waves
  const int gemmBlocks = blks(gemmWaves * 32, B);
  const int edgeBlocks = blks(EE2 * 32, B);              // one wave per edge

  for (int t = 0; t < TT; ++t) {
    const float* eat = eattr + (size_t)t * EE;
    // self-loop edge_attr = per-node mean of incoming edge_attr
    k_zero<<<blks(NN, B), B, 0, stream>>>(deg, NN);
    k_zero<<<blks(NN, B), B, 0, stream>>>(easum, NN);
    k_deg<<<blks(EE, B), B, 0, stream>>>(dst2, eat, deg, easum);
    k_ea2<<<blks(EE2, B), B, 0, stream>>>(eat, deg, easum, ea2);

    for (int conv = 0; conv < 3; ++conv) {
      const int heads = (conv < 2) ? 8 : 1;
      const int ch    = HIDC / heads;
      const bf16* Wt  = (conv == 0) ? Wt1 : (conv == 1) ? Wt2 : Wt3;
      const int Kp    = (conv == 0) ? 32 : HIDC;

      if (conv == 0)   // pad 31 -> 32 and convert timestep input
        k_cvt_pad<<<blks(NN * 32, B), B, 0, stream>>>(
            x_seq + (size_t)t * NN * 31, Abf, NN, 31, 32);
      // xp = X @ Wl + bl   (bf16 WMMA, f32 accumulate)
      k_wmma_gemm<<<gemmBlocks, B, 0, stream>>>(Abf, Wt, cv[conv].bl, xp, NN, HIDC, Kp);

      // edge attention: alpha, segment-max, exp/segment-sum, scatter aggregate
      k_zero<<<blks(NN * heads, B), B, 0, stream>>>((float*)amaxKey, NN * heads);
      k_zero<<<blks(NN * heads, B), B, 0, stream>>>(denom, NN * heads);
      k_gat_alpha<<<edgeBlocks, B, 0, stream>>>(xp, src2, dst2, ea2, cv[conv].We,
                                                cv[conv].att, alphaB, amaxKey, heads, ch);
      k_gat_soft<<<blks(EE2 * heads, B), B, 0, stream>>>(alphaB, amaxKey, dst2, denom, heads);
      k_zero<<<blks(NN * HIDC, B), B, 0, stream>>>(outb, NN * HIDC);
      k_gat_agg<<<edgeBlocks, B, 0, stream>>>(xp, alphaB, denom, src2, dst2, outb, heads, ch);

      if (conv < 2) {
        k_post<<<blks(NN * HIDC, B), B, 0, stream>>>(
            outb, cv[conv].bias, bn_g[conv], bn_b[conv], bn_m[conv], bn_v[conv], Abf);
      } else {
        k_pool<<<1, HIDC, 0, stream>>>(outb, cv[2].bias, grepr + (size_t)t * HIDC);
      }
    }
  }

  k_lstm_fc<<<1, 4 * LSTMH, 0, stream>>>(grepr, Wih0, Whh0, bih0, bhh0,
                                         Wih1, Whh1, bih1, bhh1, fcW, fcb,
                                         (float*)d_out);
}
